// LinearAE_CNN_Mamba_30253749633166
// MI455X (gfx1250) — matrix operand order repaired and checked by reading, not compiled, not run on hardware
//
#include <hip/hip_runtime.h>
#include <hip/hip_bf16.h>
#include <math.h>

// Model dims
#define B_    64
#define LSEQ  256
#define E_    128
#define HAE   2048
#define CC    256
#define NC_   10
#define L2_   128
#define DIN   512
#define DST   16
#define DTR   16
#define EL    (E_ * LSEQ)      // 32768
#define SPLITK 16

typedef __attribute__((ext_vector_type(8)))  float  v8f;
typedef __attribute__((ext_vector_type(2)))  float  v2f;
typedef __attribute__((ext_vector_type(16))) __bf16 v16bf;
typedef __attribute__((ext_vector_type(8)))  __bf16 v8bf;

// ---------------- embedding gather: e[b][l][ec] = embed[x[b][l]][ec] ----------------
__global__ void k_embed(const int* __restrict__ x, const float* __restrict__ embed,
                        float* __restrict__ e, int total) {
  int i = blockIdx.x * blockDim.x + threadIdx.x;
  if (i >= total) return;
  int ec = i % E_;
  int bl = i / E_;
  int tok = x[bl];
  e[i] = embed[(size_t)tok * E_ + ec];
}

// ---------------- fp32 WMMA GEMM, M=64 fixed (4 M-tiles per wave, weight read once) ----
// C[m][n] = sum_k A[m][k] * W[n][k].  EPI=0: write split-K partial; EPI=1: bias+ReLU.
template <int EPI>
__global__ void k_gemm_f32_m64(const float* __restrict__ A,   // 64 x K
                               const float* __restrict__ W,   // N x K
                               const float* __restrict__ bias,
                               float* __restrict__ out,       // EPI0: S x 64 x N, EPI1: 64 x N
                               int N, int K, int S) {
  int wave = (blockIdx.x * blockDim.x + threadIdx.x) >> 5;
  int ntiles = N >> 4;
  int nt = wave % ntiles;
  int s  = wave / ntiles;
  if (s >= S) return;
  int lane = threadIdx.x & 31;
  int half = lane >> 4;
  int r    = lane & 15;
  int kslice = K / S;
  int kbeg = s * kslice, kend = kbeg + kslice;

  const float* wrow = W + (size_t)(nt * 16 + r) * K;
  const float* a0r  = A + (size_t)(r)      * K;
  const float* a1r  = A + (size_t)(16 + r) * K;
  const float* a2r  = A + (size_t)(32 + r) * K;
  const float* a3r  = A + (size_t)(48 + r) * K;

  v8f acc0 = {}, acc1 = {}, acc2 = {}, acc3 = {};
#pragma unroll 4
  for (int k0 = kbeg; k0 < kend; k0 += 4) {
    int ko = k0 + 2 * half;                       // ISA 7.12.2: 32-bit A 16x4 / B 4x16 layout
    v2f b  = *(const v2f*)(wrow + ko);
    v2f a0 = *(const v2f*)(a0r + ko);
    v2f a1 = *(const v2f*)(a1r + ko);
    v2f a2 = *(const v2f*)(a2r + ko);
    v2f a3 = *(const v2f*)(a3r + ko);
    acc0 = __builtin_amdgcn_wmma_f32_16x16x4_f32(false, a0, false, b, (short)0, acc0, false, false);
    acc1 = __builtin_amdgcn_wmma_f32_16x16x4_f32(false, a1, false, b, (short)0, acc1, false, false);
    acc2 = __builtin_amdgcn_wmma_f32_16x16x4_f32(false, a2, false, b, (short)0, acc2, false, false);
    acc3 = __builtin_amdgcn_wmma_f32_16x16x4_f32(false, a3, false, b, (short)0, acc3, false, false);
  }

  v8f accs[4] = {acc0, acc1, acc2, acc3};
#pragma unroll
  for (int mt = 0; mt < 4; ++mt) {
#pragma unroll
    for (int j = 0; j < 8; ++j) {
      int m = mt * 16 + j + 8 * half;             // C/D layout: VGPR j, lane-half selects M
      int n = nt * 16 + r;
      float v = accs[mt][j];
      if (EPI == 0) {
        out[((size_t)s * 64 + m) * N + n] = v;
      } else {
        v += bias[n];
        out[(size_t)m * N + n] = v > 0.f ? v : 0.f;
      }
    }
  }
}

// ---------------- split-K reduction + bias + ReLU ----------------
__global__ void k_reduce_bias_relu(const float* __restrict__ part, const float* __restrict__ bias,
                                   float* __restrict__ out, int N, int S) {
  int i = blockIdx.x * blockDim.x + threadIdx.x;   // i over 64*N
  if (i >= 64 * N) return;
  int n = i % N;
  float acc = 0.f;
  for (int s = 0; s < S; ++s) acc += part[(size_t)s * 64 * N + i];
  acc += bias[n];
  out[i] = acc > 0.f ? acc : 0.f;
}

// ---------------- bf16 WMMA GEMM (general M): out[m][n] = sum_k A[m][k]*W[n][k] --------
__global__ void k_gemm_bf16(const __bf16* __restrict__ A,  // M x K
                            const __bf16* __restrict__ W,  // N x K
                            float* __restrict__ out,       // M x N
                            int M, int N, int K) {
  int wave = (blockIdx.x * blockDim.x + threadIdx.x) >> 5;
  int mtiles = M >> 4, ntiles = N >> 4;
  if (wave >= mtiles * ntiles) return;
  int mt = wave % mtiles;
  int nt = wave / mtiles;
  int lane = threadIdx.x & 31;
  int half = lane >> 4;
  int r    = lane & 15;

  const __bf16* arow = A + (size_t)(mt * 16 + r) * K;
  const __bf16* wrow = W + (size_t)(nt * 16 + r) * K;

  v8f acc = {};
  for (int k0 = 0; k0 < K; k0 += 32) {
    // A 16x32: lane-half selects K 0-15 / 16-31, contiguous along A row (matches B)
    v16bf a = *(const v16bf*)(arow + k0 + 16 * half);
    // B 32x16: lane-half selects K 0-15 / 16-31, contiguous along W row
    v16bf b = *(const v16bf*)(wrow + k0 + 16 * half);
    acc = __builtin_amdgcn_wmma_f32_16x16x32_bf16(false, a, false, b, (short)0, acc, false, false);
  }
#pragma unroll
  for (int j = 0; j < 8; ++j) {
    int m = mt * 16 + j + 8 * half;
    int n = nt * 16 + r;
    out[(size_t)m * N + n] = acc[j];
  }
}

// ---------------- f32 -> bf16 conversion ----------------
__global__ void k_cvt_bf16(const float* __restrict__ src, __bf16* __restrict__ dst, int n) {
  int i = blockIdx.x * blockDim.x + threadIdx.x;
  if (i >= n) return;
  dst[i] = (__bf16)src[i];
}

// ---------------- conv1d(E->C,k=5,pad2) + ReLU + maxpool2, output (B, L2, C) -----------
__global__ void k_conv_relu_pool(const float* __restrict__ h2,   // (B, LSEQ, E) = (b,l,e)
                                 const float* __restrict__ cw,   // (C, E, 5)
                                 const float* __restrict__ cb,
                                 float* __restrict__ xs,         // (B, L2, C)
                                 __bf16* __restrict__ xs_bf) {
  int i = blockIdx.x * blockDim.x + threadIdx.x;
  if (i >= B_ * L2_ * CC) return;
  int co = i % CC;
  int l2 = (i / CC) % L2_;
  int b  = i / (CC * L2_);
  float m = 0.f;   // ReLU output is >= 0, so max starts at 0 is safe after relu of both taps
  float best = -1e30f;
#pragma unroll
  for (int dl = 0; dl < 2; ++dl) {
    int l = 2 * l2 + dl;
    float acc = cb[co];
    for (int ci = 0; ci < E_; ++ci) {
      const float* wp = cw + ((size_t)co * E_ + ci) * 5;
      const float* ip = h2 + ((size_t)b * LSEQ) * E_ + ci;
#pragma unroll
      for (int t = 0; t < 5; ++t) {
        int ll = l + t - 2;
        if (ll >= 0 && ll < LSEQ) acc += ip[(size_t)ll * E_] * wp[t];
      }
    }
    acc = acc > 0.f ? acc : 0.f;
    best = acc > best ? acc : best;
  }
  m = best;
  xs[i] = m;
  xs_bf[i] = (__bf16)m;
}

// ---------------- depthwise causal conv1d(k=4) + SiLU on u half of xz ------------------
__global__ void k_dwconv_silu(const float* __restrict__ xz,   // (B, L2, 2*DIN)
                              const float* __restrict__ w,    // (DIN, 1, 4)
                              const float* __restrict__ bias,
                              float* __restrict__ u,          // (B, L2, DIN)
                              __bf16* __restrict__ u_bf) {
  int i = blockIdx.x * blockDim.x + threadIdx.x;
  if (i >= B_ * L2_ * DIN) return;
  int d = i % DIN;
  int t = (i / DIN) % L2_;
  int b = i / (DIN * L2_);
  float acc = bias[d];
#pragma unroll
  for (int j = 0; j < 4; ++j) {
    int tt = t - 3 + j;
    if (tt >= 0) acc += xz[((size_t)b * L2_ + tt) * (2 * DIN) + d] * w[d * 4 + j];
  }
  float s = acc / (1.f + __expf(-acc));   // SiLU
  u[i] = s;
  u_bf[i] = (__bf16)s;
}

// ---------------- delta = softplus(dt @ dt_proj_w.T + b) -------------------------------
__global__ void k_dt_softplus(const float* __restrict__ xdbl,  // (B*L2, 48), dt = [:,:16]
                              const float* __restrict__ dw,    // (DIN, 16)
                              const float* __restrict__ db,
                              float* __restrict__ delta) {     // (B*L2, DIN)
  int i = blockIdx.x * blockDim.x + threadIdx.x;
  if (i >= B_ * L2_ * DIN) return;
  int d  = i % DIN;
  int bt = i / DIN;
  const float* dtv = xdbl + (size_t)bt * 48;
  float acc = db[d];
#pragma unroll
  for (int r = 0; r < DTR; ++r) acc += dtv[r] * dw[d * DTR + r];
  delta[i] = acc > 20.f ? acc : log1pf(__expf(acc));
}

// ---------------- selective scan; one thread per (b,d), 16-state in registers ----------
__global__ void __launch_bounds__(DIN)
k_scan(const float* __restrict__ delta,  // (B, L2, DIN)
       const float* __restrict__ u,      // (B, L2, DIN)
       const float* __restrict__ xdbl,   // (B, L2, 48): Bm=[16:32], Cm=[32:48]
       const float* __restrict__ xz,     // (B, L2, 2*DIN): z = [DIN:]
       const float* __restrict__ A_log,  // (DIN, DST)
       const float* __restrict__ Dp,     // (DIN)
       float* __restrict__ y,            // (B, L2, DIN)
       __bf16* __restrict__ y_bf) {
  __shared__ float sB[L2_ * DST];
  __shared__ float sC[L2_ * DST];
  int b = blockIdx.x;
  int d = threadIdx.x;
  for (int i = d; i < L2_ * DST; i += DIN) {
    int t = i >> 4, n = i & 15;
    sB[i] = xdbl[((size_t)b * L2_ + t) * 48 + 16 + n];
    sC[i] = xdbl[((size_t)b * L2_ + t) * 48 + 32 + n];
  }
  __syncthreads();

  float Ad[DST], h[DST];
#pragma unroll
  for (int n = 0; n < DST; ++n) { Ad[n] = -__expf(A_log[d * DST + n]); h[n] = 0.f; }
  float Dd = Dp[d];

  for (int t = 0; t < L2_; ++t) {
    size_t idx = ((size_t)b * L2_ + t) * DIN + d;
    float dl = delta[idx];
    float uu = u[idx];
    float dbu = dl * uu;
    float yv = 0.f;
#pragma unroll
    for (int n = 0; n < DST; ++n) {
      float dA = __expf(dl * Ad[n]);
      h[n] = dA * h[n] + dbu * sB[t * DST + n];
      yv += h[n] * sC[t * DST + n];
    }
    yv += uu * Dd;
    float z = xz[((size_t)b * L2_ + t) * (2 * DIN) + DIN + d];
    yv *= z / (1.f + __expf(-z));
    y[idx] = yv;
    y_bf[idx] = (__bf16)yv;
  }
}

// ---------------- mean over time ----------------
__global__ void k_mean(const float* __restrict__ yproj,  // (B, L2, C)
                       float* __restrict__ ym) {         // (B, C)
  int i = blockIdx.x * blockDim.x + threadIdx.x;
  if (i >= B_ * CC) return;
  int c = i % CC;
  int b = i / CC;
  float acc = 0.f;
  for (int t = 0; t < L2_; ++t) acc += yproj[((size_t)b * L2_ + t) * CC + c];
  ym[i] = acc * (1.f / (float)L2_);
}

// ---------------- final FC ----------------
__global__ void k_fc(const float* __restrict__ ym, const float* __restrict__ fw,
                     const float* __restrict__ fb, float* __restrict__ out) {
  int i = blockIdx.x * blockDim.x + threadIdx.x;
  if (i >= B_ * NC_) return;
  int nc = i % NC_;
  int b  = i / NC_;
  float acc = fb[nc];
  for (int c = 0; c < CC; ++c) acc += ym[(size_t)b * CC + c] * fw[nc * CC + c];
  out[i] = acc;
}

extern "C" void kernel_launch(void* const* d_in, const int* in_sizes, int n_in,
                              void* d_out, int out_size, void* d_ws, size_t ws_size,
                              hipStream_t stream) {
  const int*   x        = (const int*)  d_in[0];
  const float* embed    = (const float*)d_in[1];
  const float* enc_w1   = (const float*)d_in[2];
  const float* enc_b1   = (const float*)d_in[3];
  const float* enc_w2   = (const float*)d_in[4];
  const float* enc_b2   = (const float*)d_in[5];
  const float* conv_w   = (const float*)d_in[6];
  const float* conv_b   = (const float*)d_in[7];
  const float* in_proj_w  = (const float*)d_in[8];
  const float* conv1d_w   = (const float*)d_in[9];
  const float* conv1d_b   = (const float*)d_in[10];
  const float* x_proj_w   = (const float*)d_in[11];
  const float* dt_proj_w  = (const float*)d_in[12];
  const float* dt_proj_b  = (const float*)d_in[13];
  const float* A_log      = (const float*)d_in[14];
  const float* Dp         = (const float*)d_in[15];
  const float* out_proj_w = (const float*)d_in[16];
  const float* fc_w       = (const float*)d_in[17];
  const float* fc_b       = (const float*)d_in[18];
  float* out = (float*)d_out;

  char* p = (char*)d_ws;
  auto alloc = [&](size_t bytes) -> char* {
    char* r = p;
    p += (bytes + 255) & ~(size_t)255;
    return r;
  };

  float*  e        = (float*) alloc((size_t)B_ * EL * 4);
  float*  part     = (float*) alloc((size_t)SPLITK * 64 * HAE * 4);
  float*  h1       = (float*) alloc((size_t)64 * HAE * 4);
  float*  h2       = (float*) alloc((size_t)64 * EL * 4);
  float*  xs       = (float*) alloc((size_t)B_ * L2_ * CC * 4);
  __bf16* xs_bf    = (__bf16*)alloc((size_t)B_ * L2_ * CC * 2);
  __bf16* inproj_bf = (__bf16*)alloc((size_t)2 * DIN * CC * 2);
  float*  xz       = (float*) alloc((size_t)B_ * L2_ * 2 * DIN * 4);
  float*  u        = (float*) alloc((size_t)B_ * L2_ * DIN * 4);
  __bf16* u_bf     = (__bf16*)alloc((size_t)B_ * L2_ * DIN * 2);
  __bf16* xproj_bf = (__bf16*)alloc((size_t)(DTR + 2 * DST) * DIN * 2);
  float*  xdbl     = (float*) alloc((size_t)B_ * L2_ * (DTR + 2 * DST) * 4);
  float*  delta    = (float*) alloc((size_t)B_ * L2_ * DIN * 4);
  float*  y        = (float*) alloc((size_t)B_ * L2_ * DIN * 4);
  __bf16* y_bf     = (__bf16*)alloc((size_t)B_ * L2_ * DIN * 2);
  __bf16* outproj_bf = (__bf16*)alloc((size_t)CC * DIN * 2);
  float*  yproj    = (float*) alloc((size_t)B_ * L2_ * CC * 4);
  float*  ym       = (float*) alloc((size_t)B_ * CC * 4);

  // 1. embedding
  {
    int total = B_ * LSEQ * E_;
    k_embed<<<(total + 255) / 256, 256, 0, stream>>>(x, embed, e, total);
  }
  // 2. GEMM1 (64x32768x2048), fp32 WMMA, split-K=16 -> partials
  {
    int waves = (HAE / 16) * SPLITK;             // 2048
    k_gemm_f32_m64<0><<<waves * 32 / 256, 256, 0, stream>>>(e, enc_w1, nullptr, part, HAE, EL, SPLITK);
  }
  // 3. reduce + bias + ReLU -> h1
  {
    int total = 64 * HAE;
    k_reduce_bias_relu<<<(total + 255) / 256, 256, 0, stream>>>(part, enc_b1, h1, HAE, SPLITK);
  }
  // 4. GEMM2 (64x2048x32768), fp32 WMMA, direct bias+ReLU -> h2
  {
    int waves = EL / 16;                          // 2048
    k_gemm_f32_m64<1><<<waves * 32 / 256, 256, 0, stream>>>(h1, enc_w2, enc_b2, h2, EL, HAE, 1);
  }
  // 5. conv + relu + maxpool -> xs (+bf16)
  {
    int total = B_ * L2_ * CC;
    k_conv_relu_pool<<<(total + 255) / 256, 256, 0, stream>>>(h2, conv_w, conv_b, xs, xs_bf);
  }
  // 6. weight conversions for bf16 GEMMs
  {
    int n = 2 * DIN * CC;
    k_cvt_bf16<<<(n + 255) / 256, 256, 0, stream>>>(in_proj_w, inproj_bf, n);
    n = (DTR + 2 * DST) * DIN;
    k_cvt_bf16<<<(n + 255) / 256, 256, 0, stream>>>(x_proj_w, xproj_bf, n);
    n = CC * DIN;
    k_cvt_bf16<<<(n + 255) / 256, 256, 0, stream>>>(out_proj_w, outproj_bf, n);
  }
  // 7. in_proj GEMM (8192x256x1024), bf16 WMMA -> xz
  {
    int waves = (B_ * L2_ / 16) * (2 * DIN / 16); // 512*64
    k_gemm_bf16<<<waves * 32 / 256, 256, 0, stream>>>(xs_bf, inproj_bf, xz, B_ * L2_, 2 * DIN, CC);
  }
  // 8. depthwise conv + SiLU -> u (+bf16)
  {
    int total = B_ * L2_ * DIN;
    k_dwconv_silu<<<(total + 255) / 256, 256, 0, stream>>>(xz, conv1d_w, conv1d_b, u, u_bf);
  }
  // 9. x_proj GEMM (8192x512x48), bf16 WMMA -> xdbl
  {
    int waves = (B_ * L2_ / 16) * ((DTR + 2 * DST) / 16); // 512*3
    k_gemm_bf16<<<waves * 32 / 256, 256, 0, stream>>>(u_bf, xproj_bf, xdbl, B_ * L2_, DTR + 2 * DST, DIN);
  }
  // 10. delta = softplus(dt @ dt_proj.T + b)
  {
    int total = B_ * L2_ * DIN;
    k_dt_softplus<<<(total + 255) / 256, 256, 0, stream>>>(xdbl, dt_proj_w, dt_proj_b, delta);
  }
  // 11. selective scan -> y (+bf16)
  k_scan<<<B_, DIN, 0, stream>>>(delta, u, xdbl, xz, A_log, Dp, y, y_bf);
  // 12. out_proj GEMM (8192x512x256), bf16 WMMA -> yproj
  {
    int waves = (B_ * L2_ / 16) * (CC / 16);      // 512*16
    k_gemm_bf16<<<waves * 32 / 256, 256, 0, stream>>>(y_bf, outproj_bf, yproj, B_ * L2_, CC, DIN);
  }
  // 13. mean over time
  {
    int total = B_ * CC;
    k_mean<<<(total + 255) / 256, 256, 0, stream>>>(yproj, ym);
  }
  // 14. final FC -> d_out
  {
    int total = B_ * NC_;
    k_fc<<<(total + 255) / 256, 256, 0, stream>>>(ym, fc_w, fc_b, out);
  }
  (void)in_sizes; (void)n_in; (void)out_size; (void)ws_size;
}
